// HyperNet_71983651881596
// MI455X (gfx1250) — matrix operand order repaired and checked by reading, not compiled, not run on hardware
//
#include <hip/hip_runtime.h>
#include <hip/hip_bf16.h>

// ---------------------------------------------------------------------------
// HyperNet fused kernel for MI455X (gfx1250, wave32, WMMA + async LDS DMA).
//
// h = (concat(keys, vgrad) - mean) / (std + eps)          [N, 4096] f32
// 4x: x = relu(h @ A_b @ B_b + bias_b)
//     h = scale_w[b, midx] * x + shift_w[b, midx] + h
// out = h[:, :2048] (flat) ++ h[:, 2048:] (flat)
//
// Strategy:
//  * prep kernel packs A/B weights once into fragment-native f16 in d_ws
//    (each WMMA B-operand = one aligned 32B load, no per-use cvt)
//  * one workgroup = 16 rows of h resident in LDS across all 4 blocks
//  * h tile ingested via GLOBAL_LOAD_ASYNC_TO_LDS_B128 (ASYNCcnt-tracked DMA,
//    no VGPR staging), then normalized in place
//  * GEMMs via v_wmma_f32_16x16x32_f16, f32 accumulate
//  * all activation HBM traffic exactly once: 268 MB in, 268 MB out (~23 us
//    at 23.3 TB/s; compute is negligible on the WMMA pipe)
// ---------------------------------------------------------------------------

#define KEY_SIZE   2048
#define SIZE       4096
#define RANK       32
#define N_BLOCKS   4
#define N_MODULES  64
#define ROWS       16          // rows per workgroup
#define THREADS    256         // 8 wave32s
#define WAVES      8
#define EPS_F32    1.1920928955078125e-07f

#define KCHUNKS    (SIZE / 32)            // 128 K-chunks for stage 1
#define CTILES     (SIZE / 16)            // 256 column tiles for stage 2
// packed f16 weights: [b][kchunk][nb][lane][e] and [b][ctile][lane][e]
#define APACK_PER_B (KCHUNKS * 2 * 32 * 16)   // 131072 f16 per block
#define BPACK_PER_B (CTILES * 32 * 16)        // 131072 f16 per block
#define APACK_ELEMS (N_BLOCKS * APACK_PER_B)  // 524288 (1 MB)
#define BPACK_ELEMS (N_BLOCKS * BPACK_PER_B)  // 524288 (1 MB)

typedef __attribute__((ext_vector_type(16))) _Float16 v16h;
typedef __attribute__((ext_vector_type(8)))  float    v8f;

__device__ __forceinline__ v16h cvt16(float4 a0, float4 a1, float4 a2, float4 a3) {
    v16h r;
    r[0]  = (_Float16)a0.x; r[1]  = (_Float16)a0.y;
    r[2]  = (_Float16)a0.z; r[3]  = (_Float16)a0.w;
    r[4]  = (_Float16)a1.x; r[5]  = (_Float16)a1.y;
    r[6]  = (_Float16)a1.z; r[7]  = (_Float16)a1.w;
    r[8]  = (_Float16)a2.x; r[9]  = (_Float16)a2.y;
    r[10] = (_Float16)a2.z; r[11] = (_Float16)a2.w;
    r[12] = (_Float16)a3.x; r[13] = (_Float16)a3.y;
    r[14] = (_Float16)a3.z; r[15] = (_Float16)a3.w;
    return r;
}

// Async DMA: global -> LDS, 16 bytes per lane, tracked by ASYNCcnt.
// ldsa = workgroup-relative LDS byte offset (low 32 bits of generic pointer).
__device__ __forceinline__ void async_copy_b128(unsigned int ldsa, const void* gsrc) {
    asm volatile("global_load_async_to_lds_b128 %0, %1, off"
                 :: "v"(ldsa), "v"((unsigned long long)(uintptr_t)gsrc)
                 : "memory");
}
__device__ __forceinline__ void wait_asynccnt0() {
    asm volatile("s_wait_asynccnt 0x0" ::: "memory");
}

// ---------------------------------------------------------------------------
// One-shot weight pack: f32 -> f16 in WMMA-fragment-native order.
// 16-bit B-fragment layout (ISA 7.12.2): lane = hi*16+l, N = l (+nb*16),
// element e -> K = (chunk base) + (e>>3)*16 + hi*8 + (e&7).
// ---------------------------------------------------------------------------
__global__ __launch_bounds__(THREADS)
void pack_weights_f16(const float* __restrict__ Aw,   // [4, SIZE, RANK]
                      const float* __restrict__ Bw,   // [4, RANK, SIZE]
                      _Float16* __restrict__ apack,
                      _Float16* __restrict__ bpack)
{
    const int i = blockIdx.x * THREADS + threadIdx.x;
    if (i < APACK_ELEMS) {
        const int e    = i & 15;
        const int lane = (i >> 4) & 31;
        const int nb   = (i >> 9) & 1;
        const int kc   = (i >> 10) & (KCHUNKS - 1);
        const int b    = i >> 17;
        const int hi = lane >> 4, l = lane & 15;
        const int k = kc * 32 + (e >> 3) * 16 + hi * 8 + (e & 7);
        const int r = nb * 16 + l;
        apack[i] = (_Float16)Aw[((size_t)b * SIZE + k) * RANK + r];
    }
    if (i < BPACK_ELEMS) {
        const int e    = i & 15;
        const int lane = (i >> 4) & 31;
        const int ct   = (i >> 9) & (CTILES - 1);
        const int b    = i >> 17;
        const int hi = lane >> 4, l = lane & 15;
        const int k   = (e >> 3) * 16 + hi * 8 + (e & 7);
        const int col = ct * 16 + l;
        bpack[i] = (_Float16)Bw[((size_t)b * RANK + k) * SIZE + col];
    }
}

__global__ __launch_bounds__(THREADS)
void hypernet_fused_kernel(const float* __restrict__ keys,
                           const float* __restrict__ vgrad,
                           const int*   __restrict__ module_idx,
                           const float* __restrict__ mean,
                           const float* __restrict__ stdv,
                           const _Float16* __restrict__ apack,
                           const _Float16* __restrict__ bpack,
                           const float* __restrict__ biasw,   // [4, SIZE]
                           const float* __restrict__ scale_w, // [4, 64, SIZE]
                           const float* __restrict__ shift_w, // [4, 64, SIZE]
                           float* __restrict__ out,           // two flat halves
                           int n_rows)
{
    extern __shared__ float smem[];
    float* hbuf  = smem;                        // ROWS * SIZE f32   (256 KB)
    float* Tbuf  = smem + ROWS * SIZE;          // ROWS * RANK f32   (2 KB)
    int*   midxl = (int*)(Tbuf + ROWS * RANK);  // ROWS ints

    const int tid  = threadIdx.x;
    const int wave = tid >> 5;
    const int lane = tid & 31;
    const int l    = lane & 15;   // N / M-row lane within half
    const int hi   = lane >> 4;   // lane half (K / M split per ISA layouts)
    const int row0 = blockIdx.x * ROWS;

    // ---------------- Stage 0a: async DMA raw tile into LDS -----------------
    for (int i = tid * 4; i < ROWS * SIZE; i += THREADS * 4) {
        const int m   = i >> 12;
        const int col = i & (SIZE - 1);
        const int row = row0 + m;
        const float* src = (col < KEY_SIZE)
            ? &keys[(size_t)row * KEY_SIZE + col]
            : &vgrad[(size_t)row * KEY_SIZE + (col - KEY_SIZE)];
        async_copy_b128((unsigned int)(uintptr_t)&hbuf[i], src);
    }
    if (tid < ROWS) midxl[tid] = module_idx[row0 + tid];
    wait_asynccnt0();
    __syncthreads();

    // ---------------- Stage 0b: normalize in place --------------------------
    for (int i = tid * 4; i < ROWS * SIZE; i += THREADS * 4) {
        const int col = i & (SIZE - 1);
        const float4 mv = *(const float4*)&mean[col];
        const float4 sv = *(const float4*)&stdv[col];
        float4 v = *(const float4*)&hbuf[i];
        v.x = (v.x - mv.x) / (sv.x + EPS_F32);
        v.y = (v.y - mv.y) / (sv.y + EPS_F32);
        v.z = (v.z - mv.z) / (sv.z + EPS_F32);
        v.w = (v.w - mv.w) / (sv.w + EPS_F32);
        *(float4*)&hbuf[i] = v;
    }
    __syncthreads();

    // ---------------- 4 residual low-rank blocks ----------------------------
    for (int b = 0; b < N_BLOCKS; ++b) {
        const _Float16* Apb   = apack   + (size_t)b * APACK_PER_B;
        const _Float16* Bpb   = bpack   + (size_t)b * BPACK_PER_B;
        const float*    biasb = biasw   + (size_t)b * SIZE;
        const float*    swb   = scale_w + (size_t)b * N_MODULES * SIZE;
        const float*    shb   = shift_w + (size_t)b * N_MODULES * SIZE;

        for (int i = tid; i < ROWS * RANK; i += THREADS) Tbuf[i] = 0.0f;
        __syncthreads();

        // ---- Stage 1: T[16,32] = h[16,4096] @ A_b[4096,32] ----
        // each wave owns a 512-wide K slice; one A-fragment feeds 2 WMMAs
        // (both rank halves); partials reduced with ds_add_f32.
        {
            v8f acc0 = {}, acc1 = {};
            for (int kc = 0; kc < KCHUNKS / WAVES; ++kc) {
                const int kcg = wave * (KCHUNKS / WAVES) + kc;   // 0..127
                const int kb  = kcg * 32;
                // A-operand from LDS: K runs [kb+hi*8, +8) and [kb+16+hi*8, +8)
                const float4* hp = (const float4*)&hbuf[l * SIZE + kb + hi * 8];
                const v16h af = cvt16(hp[0], hp[1], hp[4], hp[5]);
                // B-operand: packed f16, one 32B load per rank half
                const v16h bf0 = *(const v16h*)(Apb + ((size_t)kcg * 2 + 0) * 512 + lane * 16);
                const v16h bf1 = *(const v16h*)(Apb + ((size_t)kcg * 2 + 1) * 512 + lane * 16);
                acc0 = __builtin_amdgcn_wmma_f32_16x16x32_f16(
                           false, af, false, bf0, (short)0, acc0, false, false);
                acc1 = __builtin_amdgcn_wmma_f32_16x16x32_f16(
                           false, af, false, bf1, (short)0, acc1, false, false);
            }
            // C/D layout: VGPR g -> M = g + hi*8, N = l (+16 for second half)
            #pragma unroll
            for (int g = 0; g < 8; ++g) {
                atomicAdd(&Tbuf[(g + hi * 8) * RANK + l],      acc0[g]);
                atomicAdd(&Tbuf[(g + hi * 8) * RANK + 16 + l], acc1[g]);
            }
        }
        __syncthreads();

        // ---- Stage 2: x = T[16,32] @ B_b[32,4096]; fused epilogue ----
        {
            const float4* tp = (const float4*)&Tbuf[l * RANK + hi * 8];
            const v16h ta = cvt16(tp[0], tp[1], tp[4], tp[5]);
            for (int ct = wave; ct < CTILES; ct += WAVES) {
                const v16h bf = *(const v16h*)(Bpb + (size_t)ct * 512 + lane * 16);
                v8f c = {};
                c = __builtin_amdgcn_wmma_f32_16x16x32_f16(
                        false, ta, false, bf, (short)0, c, false, false);

                const int col  = ct * 16 + l;
                const float bi = biasb[col];
                #pragma unroll
                for (int g = 0; g < 8; ++g) {
                    const int m  = g + hi * 8;
                    const int mi = midxl[m];
                    float x = c[g] + bi;
                    x = fmaxf(x, 0.0f);
                    const size_t eoff = (size_t)mi * SIZE + col;
                    float y = swb[eoff] * x + shb[eoff] + hbuf[m * SIZE + col];
                    hbuf[m * SIZE + col] = y;
                }
            }
        }
        __syncthreads();
    }

    // ---------------- Stage 3: write out (two flat halves, b128) ------------
    const size_t vals_base = (size_t)n_rows * KEY_SIZE;
    for (int i = tid * 4; i < ROWS * SIZE; i += THREADS * 4) {
        const int m   = i >> 12;
        const int col = i & (SIZE - 1);
        const int row = row0 + m;
        const float4 v = *(const float4*)&hbuf[i];
        if (col < KEY_SIZE)
            *(float4*)&out[(size_t)row * KEY_SIZE + col] = v;
        else
            *(float4*)&out[vals_base + (size_t)row * KEY_SIZE + (col - KEY_SIZE)] = v;
    }
}

extern "C" void kernel_launch(void* const* d_in, const int* in_sizes, int n_in,
                              void* d_out, int out_size, void* d_ws, size_t ws_size,
                              hipStream_t stream) {
    (void)n_in; (void)out_size; (void)ws_size;

    const float* keys    = (const float*)d_in[0];
    const float* vgrad   = (const float*)d_in[1];
    const int*   midx    = (const int*)  d_in[2];
    const float* mean    = (const float*)d_in[3];
    const float* stdv    = (const float*)d_in[4];
    const float* Aw      = (const float*)d_in[5];
    const float* Bw      = (const float*)d_in[6];
    const float* biasw   = (const float*)d_in[7];
    const float* scale_w = (const float*)d_in[8];
    const float* shift_w = (const float*)d_in[9];
    float*       out     = (float*)d_out;

    _Float16* apack = (_Float16*)d_ws;                 // 1 MB
    _Float16* bpack = apack + APACK_ELEMS;             // 1 MB (ws >= 2 MB)

    // one-shot (re-run every call for determinism; ~2 MB, negligible)
    pack_weights_f16<<<dim3(APACK_ELEMS / THREADS), dim3(THREADS), 0, stream>>>(
        Aw, Bw, apack, bpack);

    const int n_rows = in_sizes[0] / KEY_SIZE;   // 16384
    const int n_wg   = n_rows / ROWS;            // 1024

    const size_t lds_bytes = (size_t)(ROWS * SIZE + ROWS * RANK) * sizeof(float)
                           + (size_t)ROWS * sizeof(int);   // ~264 KB (< 320 KB/WG)

    hypernet_fused_kernel<<<dim3(n_wg), dim3(THREADS), lds_bytes, stream>>>(
        keys, vgrad, midx, mean, stdv, apack, bpack, biasw, scale_w, shift_w,
        out, n_rows);
}
